// MedicalStyleGenerator_5394478924600
// MI455X (gfx1250) — matrix operand order repaired and checked by reading, not compiled, not run on hardware
//
#include <hip/hip_runtime.h>
#include <hip/hip_bf16.h>
#include <stdint.h>

typedef __attribute__((ext_vector_type(16))) _Float16 v16h;
typedef __attribute__((ext_vector_type(8)))  float    v8f;

constexpr int BATCH = 16;

// ---------------- mapping MLP: style = MLP(latent||cond) ----------------
__global__ void mapping_kernel(const float* __restrict__ latent,
                               const float* __restrict__ cond,
                               const float* __restrict__ W0, const float* __restrict__ b0,
                               const float* __restrict__ W1, const float* __restrict__ b1,
                               const float* __restrict__ W2, const float* __restrict__ b2,
                               const float* __restrict__ W3, const float* __restrict__ b3,
                               float* __restrict__ style) {
  __shared__ float sbuf[BATCH][520];
  int t = threadIdx.x; // 512 threads
  for (int i = t; i < BATCH * 518; i += 512) {
    int b = i / 518, k = i % 518;
    sbuf[b][k] = (k < 512) ? latent[b * 512 + k] : cond[b * 6 + (k - 512)];
  }
  __syncthreads();
  const float* Ws[4] = {W0, W1, W2, W3};
  const float* bs[4] = {b0, b1, b2, b3};
  int o = t; // one output column per thread, all batches
  for (int l = 0; l < 4; ++l) {
    int fan = (l == 0) ? 518 : 512;
    float acc[BATCH];
    #pragma unroll
    for (int b = 0; b < BATCH; ++b) acc[b] = bs[l][o];
    const float* Wl = Ws[l];
    for (int k = 0; k < fan; ++k) {
      float wv = Wl[k * 512 + o];
      #pragma unroll
      for (int b = 0; b < BATCH; ++b) acc[b] += sbuf[b][k] * wv;
    }
    __syncthreads();
    if (l < 3) {
      #pragma unroll
      for (int b = 0; b < BATCH; ++b) {
        float v = acc[b];
        sbuf[b][o] = (v > 0.f) ? v : 0.2f * v;
      }
    } else {
      #pragma unroll
      for (int b = 0; b < BATCH; ++b) style[b * 512 + o] = acc[b];
    }
    __syncthreads();
  }
}

// ---------------- per-block style projection: smod = 1 + style@sw + sb ----------------
__global__ void styleproj_kernel(const float* __restrict__ style,
                                 const float* __restrict__ sw, const float* __restrict__ sb,
                                 float* __restrict__ smod, int C) {
  int t = blockIdx.x * blockDim.x + threadIdx.x;
  if (t >= BATCH * C) return;
  int b = t / C, c = t % C;
  float acc = sb[c];
  const float* st = style + b * 512;
  for (int k = 0; k < 512; ++k) acc += st[k] * sw[k * C + c];
  smod[t] = 1.0f + acc;
}

__global__ void f32_to_f16_kernel(const float* __restrict__ in, _Float16* __restrict__ out, int n) {
  int t = blockIdx.x * blockDim.x + threadIdx.x;
  if (t < n) out[t] = (_Float16)in[t];
}

// ---------------- pack conv weights into WMMA A-fragment layout ----------------
// packed[((ct*ksteps+ks)*32 + lane)*16 + j] : lane = m + 16*((k>>3)&1), j = (k&7)|(k>=16?8:0)
__global__ void packA_kernel(const float* __restrict__ wgt, _Float16* __restrict__ packed,
                             int Cin, int Cout, int coTiles, int ksteps) {
  int idx = blockIdx.x * blockDim.x + threadIdx.x;
  int total = coTiles * ksteps * 512;
  if (idx >= total) return;
  int j = idx & 15;
  int lane = (idx >> 4) & 31;
  int cks = idx >> 9;
  int ks = cks % ksteps, ct = cks / ksteps;
  int m = lane & 15;
  int k = (j & 7) | ((lane >> 4) << 3) | ((j >> 3) << 4);
  int kg = ks * 32 + k;
  int cog = ct * 16 + m;
  float wv = 0.f;
  if (kg < Cin * 9 && cog < Cout) {
    int c = kg / 9, r = kg % 9;
    wv = wgt[((cog * Cin + c) * 3 + (r / 3)) * 3 + (r % 3)];
  }
  packed[idx] = (_Float16)wv;
}

// ---------------- fused (upsample) + modulate + conv3x3 + noise + lrelu, WMMA implicit GEMM ----------------
// grid: (npix/(64*NT), coTiles/MT, BATCH), block 128 (4 waves).
// Each wave: (16*MT) out-channels x (16*NT) pixels. One B stage feeds NT*MT WMMAs;
// MT A fragments come straight from packed global weights (no LDS for A, no barriers).
// Register budget kept near acc<=32 VGPRs to avoid scratch spills (seen at NT=4,MT=2).
template <bool UPSAMPLE, int NT, int MT>
__global__ void conv3x3_wmma_kernel(const _Float16* __restrict__ xin, int xBatchStride,
                                    const _Float16* __restrict__ packedA,
                                    const float* __restrict__ bias,
                                    const float* __restrict__ smod,
                                    const float* __restrict__ noise,
                                    const float* __restrict__ nscale,
                                    _Float16* __restrict__ out,
                                    int Cin, int Cout, int Hin, int hshift) {
  const int Hout = UPSAMPLE ? (Hin * 2) : Hin;
  const int npix = Hout * Hout;
  const int hmask = Hout - 1;
  const int wid = threadIdx.x >> 5;
  const int lane = threadIdx.x & 31;
  const int b = blockIdx.z;
  const int coBase = blockIdx.y * (16 * MT);
  const int pixBase = (blockIdx.x * 4 + wid) * (16 * NT);
  const int Ktot = Cin * 9;
  const int ksteps = (Ktot + 31) >> 5;

  // per-wave B tile, 17-dword row pitch -> conflict-free 16-row gathers
  __shared__ alignas(16) _Float16 lsB[4][16 * NT][34];
  __shared__ int   upI[UPSAMPLE ? 512 : 1];
  __shared__ float upF[UPSAMPLE ? 512 : 1];
  if constexpr (UPSAMPLE) {
    float scale = (float)(Hin - 1) / (float)(2 * Hin - 1);
    for (int u = threadIdx.x; u < Hout; u += 128) {
      float pos = u * scale;
      int i0 = (int)pos; if (i0 > Hin - 2) i0 = Hin - 2;
      upI[u] = i0; upF[u] = pos - (float)i0;
    }
    __syncthreads();
  }

  v8f acc[NT * MT];
  #pragma unroll
  for (int t = 0; t < NT * MT; ++t) acc[t] = (v8f){};

  for (int ks = 0; ks < ksteps; ++ks) {
    // ---- MT A fragments: coalesced 32B loads from pre-packed weights ----
    union frag { uint32_t u[8]; v16h v; };
    frag fa[MT];
    #pragma unroll
    for (int m = 0; m < MT; ++m) {
      const _Float16* pa = packedA +
          ((((size_t)blockIdx.y * MT + m) * ksteps + ks) * 32 + lane) * 16;
      fa[m].v = *(const v16h*)pa;
      if (ks + 1 < ksteps) __builtin_prefetch(pa + 512, 0, 3);
    }

    // ---- stage B (transposed [pixel][k]): modulated (+bilinear) im2col, per-wave ----
    int kg = ks * 32 + lane;
    bool kvalid = (kg < Ktot);
    int dy = 0, dx = 0;
    float sm = 0.f;
    const _Float16* xc = xin;
    if (kvalid) {
      int c = kg / 9, r = kg % 9;
      dy = r / 3 - 1; dx = r % 3 - 1;
      sm = smod[b * Cin + c];
      xc = xin + (size_t)b * xBatchStride + (size_t)c * Hin * Hin;
    }
    for (int n = 0; n < 16 * NT; ++n) {
      float v = 0.f;
      if (kvalid) {
        int p = pixBase + n;
        int py = p >> hshift, px = p & hmask;
        int uy = py + dy, ux = px + dx;
        if (uy >= 0 && uy < Hout && ux >= 0 && ux < Hout) {
          if constexpr (UPSAMPLE) {
            int iy0 = upI[uy]; float fy = upF[uy];
            int ix0 = upI[ux]; float fx = upF[ux];
            const _Float16* p00 = xc + iy0 * Hin + ix0;
            float x00 = (float)p00[0],   x01 = (float)p00[1];
            float x10 = (float)p00[Hin], x11 = (float)p00[Hin + 1];
            v = (x00 * (1.f - fy) + x10 * fy) * (1.f - fx)
              + (x01 * (1.f - fy) + x11 * fy) * fx;
          } else {
            v = (float)xc[uy * Hin + ux];
          }
          v *= sm;
        }
      }
      lsB[wid][n][lane] = (_Float16)v;
    }

    // ---- NT*MT WMMAs per staged B tile; same-wave in-order LDS => race-free ----
    int mn = lane & 15;
    int hiB = (lane >= 16) ? 8 : 0;
    #pragma unroll
    for (int t = 0; t < NT; ++t) {
      frag fb;
      #pragma unroll
      for (int r = 0; r < 8; ++r)
        fb.u[r] = *(const uint32_t*)&lsB[wid][t * 16 + mn][hiB + 2 * r + ((r >= 4) ? 8 : 0)];
      #pragma unroll
      for (int m = 0; m < MT; ++m)
        acc[t * MT + m] = __builtin_amdgcn_wmma_f32_16x16x32_f16(
            false, fa[m].v, false, fb.v, (short)0, acc[t * MT + m], false, false);
    }
  }

  // ---- epilogue: + bias + noise*scale, leaky-relu, f16 store ----
  float ns = nscale[0];
  int mn = lane & 15;
  int rowOff = (lane >= 16) ? 8 : 0;
  #pragma unroll
  for (int t = 0; t < NT; ++t) {
    int p = pixBase + t * 16 + mn;
    float nz = noise[b * npix + p] * ns;
    #pragma unroll
    for (int m = 0; m < MT; ++m) {
      #pragma unroll
      for (int r = 0; r < 8; ++r) {
        int co = coBase + m * 16 + r + rowOff;
        if (co < Cout) {
          float v = acc[t * MT + m][r] + bias[co] + nz;
          v = (v > 0.f) ? v : 0.2f * v;
          out[((size_t)b * Cout + co) * npix + p] = (_Float16)v;
        }
      }
    }
  }
}

// ---------------- 1x1 conv to RGB + tanh ----------------
__global__ void rgb_tanh_kernel(const _Float16* __restrict__ x,
                                const float* __restrict__ rgbw,
                                const float* __restrict__ rgbb,
                                float* __restrict__ out, int Cin, int npix) {
  int t = blockIdx.x * blockDim.x + threadIdx.x;
  int total = BATCH * npix;
  if (t >= total) return;
  int b = t / npix, p = t % npix;
  float sum = rgbb[0];
  for (int c = 0; c < Cin; ++c)
    sum += (float)x[((size_t)b * Cin + c) * npix + p] * rgbw[c];
  out[t] = tanhf(sum);
}

extern "C" void kernel_launch(void* const* d_in, const int* in_sizes, int n_in,
                              void* d_out, int out_size, void* d_ws, size_t ws_size,
                              hipStream_t stream) {
  (void)in_sizes; (void)n_in; (void)out_size; (void)ws_size;
  const int CHS[7] = {512, 256, 128, 64, 32, 16, 8};

  const float* latent = (const float*)d_in[0];
  const float* cond   = (const float*)d_in[1];
  const float* mapW[4]; const float* mapB[4];
  for (int i = 0; i < 4; ++i) { mapW[i] = (const float*)d_in[2 + 2 * i]; mapB[i] = (const float*)d_in[3 + 2 * i]; }
  const float* cst  = (const float*)d_in[10];
  const float* rgbw = (const float*)d_in[81];
  const float* rgbb = (const float*)d_in[82];

  char* ws = (char*)d_ws;
  float*    style = (float*)ws;                 // 32 KB
  float*    smod1 = (float*)(ws + 32768);       // 32 KB
  float*    smod2 = (float*)(ws + 65536);       // 32 KB
  _Float16* cst16 = (_Float16*)(ws + 98304);    // 16 KB
  size_t bufEl = (size_t)BATCH * 8 * 512 * 512; // 33.5M elems (67 MB f16)
  _Float16* bufA = (_Float16*)(ws + 131072);
  _Float16* bufB = bufA + bufEl;
  _Float16* packBase = bufB + bufEl;            // ~15 MB of packed weights
  size_t packOff = 0;

  mapping_kernel<<<1, 512, 0, stream>>>(latent, cond, mapW[0], mapB[0], mapW[1], mapB[1],
                                        mapW[2], mapB[2], mapW[3], mapB[3], style);
  f32_to_f16_kernel<<<(8192 + 255) / 256, 256, 0, stream>>>(cst, cst16, 8192);

  const _Float16* xin = cst16;
  int xStride = 0; // const is broadcast over batch
  int Cin = 512;
  for (int i = 0; i < 7; ++i) {
    int Cout = CHS[i];
    int Hout = 8 << i;
    int Hin  = Hout >> 1;
    int npix = Hout * Hout;
    int hshift = 3 + i;
    const float* w1  = (const float*)d_in[11 + 10 * i + 0];
    const float* b1  = (const float*)d_in[11 + 10 * i + 1];
    const float* w2  = (const float*)d_in[11 + 10 * i + 2];
    const float* b2  = (const float*)d_in[11 + 10 * i + 3];
    const float* sw1 = (const float*)d_in[11 + 10 * i + 4];
    const float* sb1 = (const float*)d_in[11 + 10 * i + 5];
    const float* sw2 = (const float*)d_in[11 + 10 * i + 6];
    const float* sb2 = (const float*)d_in[11 + 10 * i + 7];
    const float* n1s = (const float*)d_in[11 + 10 * i + 8];
    const float* n2s = (const float*)d_in[11 + 10 * i + 9];
    const float* nz1 = (const float*)d_in[83 + 2 * i];
    const float* nz2 = (const float*)d_in[84 + 2 * i];

    int n1 = BATCH * Cin;
    styleproj_kernel<<<(n1 + 127) / 128, 128, 0, stream>>>(style, sw1, sb1, smod1, Cin);
    int n2 = BATCH * Cout;
    styleproj_kernel<<<(n2 + 127) / 128, 128, 0, stream>>>(style, sw2, sb2, smod2, Cout);

    int coTiles = (Cout + 15) / 16;
    int ks1 = (Cin * 9 + 31) / 32;
    int ks2 = (Cout * 9 + 31) / 32;
    _Float16* pk1 = packBase + packOff; packOff += (size_t)coTiles * ks1 * 512;
    _Float16* pk2 = packBase + packOff; packOff += (size_t)coTiles * ks2 * 512;
    {
      int tot = coTiles * ks1 * 512;
      packA_kernel<<<(tot + 255) / 256, 256, 0, stream>>>(w1, pk1, Cin, Cout, coTiles, ks1);
    }
    {
      int tot = coTiles * ks2 * 512;
      packA_kernel<<<(tot + 255) / 256, 256, 0, stream>>>(w2, pk2, Cout, Cout, coTiles, ks2);
    }

    if (npix >= 256 && Cout >= 32) {          // blocks 1-4: NT=2, MT=2 (acc=32 regs, no spills)
      dim3 grd(npix / 128, coTiles / 2, BATCH);
      conv3x3_wmma_kernel<true, 2, 2><<<grd, 128, 0, stream>>>(xin, xStride, pk1, b1, smod1, nz1, n1s,
                                                               bufB, Cin, Cout, Hin, hshift);
      conv3x3_wmma_kernel<false, 2, 2><<<grd, 128, 0, stream>>>(bufB, Cout * npix, pk2, b2, smod2, nz2, n2s,
                                                                bufA, Cout, Cout, Hout, hshift);
    } else if (npix >= 256) {                 // blocks 5-6: NT=4, MT=1 (single co-tile)
      dim3 grd(npix / 256, coTiles, BATCH);
      conv3x3_wmma_kernel<true, 4, 1><<<grd, 128, 0, stream>>>(xin, xStride, pk1, b1, smod1, nz1, n1s,
                                                               bufB, Cin, Cout, Hin, hshift);
      conv3x3_wmma_kernel<false, 4, 1><<<grd, 128, 0, stream>>>(bufB, Cout * npix, pk2, b2, smod2, nz2, n2s,
                                                                bufA, Cout, Cout, Hout, hshift);
    } else {                                   // block 0: 64 pixels, NT=1, MT=2
      dim3 grd(npix / 64, coTiles / 2, BATCH);
      conv3x3_wmma_kernel<true, 1, 2><<<grd, 128, 0, stream>>>(xin, xStride, pk1, b1, smod1, nz1, n1s,
                                                               bufB, Cin, Cout, Hin, hshift);
      conv3x3_wmma_kernel<false, 1, 2><<<grd, 128, 0, stream>>>(bufB, Cout * npix, pk2, b2, smod2, nz2, n2s,
                                                                bufA, Cout, Cout, Hout, hshift);
    }
    xin = bufA; xStride = Cout * npix;
    Cin = Cout;
  }

  int npix = 512 * 512;
  int total = BATCH * npix;
  rgb_tanh_kernel<<<(total + 255) / 256, 256, 0, stream>>>(bufA, rgbw, rgbb, (float*)d_out, 8, npix);
}
